// Markonv_3015067041992
// MI455X (gfx1250) — matrix operand order repaired and checked by reading, not compiled, not run on hardware
//
#include <hip/hip_runtime.h>

typedef __attribute__((ext_vector_type(16))) __bf16 v16bf;
typedef __attribute__((ext_vector_type(8)))  float  v8f;

#define KLEN   16
#define CH     4
#define NK     128            // number of kernels (K of GEMM M-dim)
#define LLEN   10000
#define RED    256            // KLEN*CH*CH reduction length
#define POUT   9984           // LLEN - KLEN
#define PTILES 624            // POUT / 16
#define PT_PER_BLOCK 48       // 13 blocks per batch in y
#define NWAVES 8              // 256 threads

// round-to-nearest-even f32 -> bf16 (bit trick, used in LDS staging)
static __device__ __forceinline__ unsigned short f2bf_rne(float f) {
  unsigned int u = __builtin_bit_cast(unsigned int, f);
  unsigned int r = u + 0x7FFFu + ((u >> 16) & 1u);
  return (unsigned short)(r >> 16);
}

__global__ __launch_bounds__(256) void markonv_wmma_bf16(
    const float* __restrict__ x,      // [B, L, C]
    const float* __restrict__ kern,   // [KLEN, C, C, NK] -> kern[r*NK + k]
    float* __restrict__ out) {        // [B, NK, POUT]
  // W fragments for ALL 8 k-tiles, pre-swizzled into WMMA A layout.
  // ldsW[ktile][j][lane][dw]  (dw = packed bf16 pair), 8*8*32*8 dwords = 64KB
  __shared__ unsigned int ldsW[8 * 8 * 32 * 8];

  const int tid = threadIdx.x;

  // ---- cooperative staging: kernel f32 -> bf16 A-fragments in LDS ----
  for (int idx = tid; idx < 8 * 8 * 32 * 8; idx += 256) {
    int dw   = idx & 7;
    int lane = (idx >> 3) & 31;
    int j    = (idx >> 8) & 7;
    int kt   = idx >> 11;
    int hi   = lane >> 4;                  // A layout: lane half selects K range
    int kg   = kt * 16 + (lane & 15);      // global kernel index (M)
    int kpair = 2 * dw + hi * 16;
    int r0 = 32 * j + kpair;               // reduction index
    float w0 = kern[r0 * NK + kg];
    float w1 = kern[(r0 + 1) * NK + kg];
    ldsW[idx] = (unsigned int)f2bf_rne(w0) | ((unsigned int)f2bf_rne(w1) << 16);
  }
  __syncthreads();

  const int wave = tid >> 5;
  const int lane = tid & 31;
  const int hi   = lane >> 4;              // B layout: high lanes hold K=16..31 (i = 2j+1)
  const int n    = lane & 15;              // output column within tile
  const int b    = blockIdx.x;
  const int pg   = blockIdx.y;

  const float4* __restrict__ x4 =
      reinterpret_cast<const float4*>(x) + (size_t)b * LLEN;  // x[b, t, 0..3]

  for (int it = 0; it < PT_PER_BLOCK / NWAVES; ++it) {
    int ptile = pg * PT_PER_BLOCK + it * NWAVES + wave;   // uniform per wave
    if (ptile >= PTILES) break;
    int p0 = ptile * 16;

    v8f acc[8];
    v8f zero = {};
#pragma unroll
    for (int kt = 0; kt < 8; ++kt) acc[kt] = zero;

#pragma unroll
    for (int j = 0; j < 8; ++j) {
      // Build B fragment: lane holds column p = p0+n, K rows (hi?16:0)+0..15.
      // K row = i*16 + c1*4 + c2 with i = 2j + hi  ->  x[p+i, c1] * x[p+i+1, c2]
      int i = 2 * j + hi;
      float4 va = x4[p0 + n + i];
      float4 vb = x4[p0 + n + i + 1];
      const float* xa = reinterpret_cast<const float*>(&va);
      const float* xb = reinterpret_cast<const float*>(&vb);
      v16bf bmat;
#pragma unroll
      for (int c = 0; c < 16; ++c)
        bmat[c] = (__bf16)(xa[c >> 2] * xb[c & 3]);

      // One B fragment feeds 8 WMMAs (all k-tiles): WMMA-bound inner loop.
#pragma unroll
      for (int kt = 0; kt < 8; ++kt) {
        const v16bf amat = *reinterpret_cast<const v16bf*>(
            &ldsW[(((kt * 8 + j) * 32) + lane) * 8]);
        acc[kt] = __builtin_amdgcn_wmma_f32_16x16x32_bf16(
            false, amat, false, bmat, (short)0, acc[kt], false, false);
      }
    }

    // C/D layout: VGPR v, low lanes -> M=v, high lanes -> M=8+v; N = lane&15
#pragma unroll
    for (int kt = 0; kt < 8; ++kt) {
#pragma unroll
      for (int v = 0; v < 8; ++v) {
        int k = kt * 16 + hi * 8 + v;
        out[((size_t)b * NK + k) * POUT + p0 + n] = acc[kt][v];
      }
    }
  }
}

extern "C" void kernel_launch(void* const* d_in, const int* in_sizes, int n_in,
                              void* d_out, int out_size, void* d_ws, size_t ws_size,
                              hipStream_t stream) {
  const float* x    = (const float*)d_in[0];
  const float* kern = (const float*)d_in[1];
  float* out        = (float*)d_out;
  int B = in_sizes[0] / (LLEN * CH);                     // 128
  dim3 grid(B, (PTILES + PT_PER_BLOCK - 1) / PT_PER_BLOCK, 1);  // (128, 13)
  markonv_wmma_bf16<<<grid, 256, 0, stream>>>(x, kern, out);
}
